// FrequencySparseAttention_24567212933768
// MI455X (gfx1250) — compile-verified, matrix operand order checked by reading, not run on hardware
//
#include <hip/hip_runtime.h>
#include <hip/hip_bf16.h>

typedef _Float16 half16 __attribute__((ext_vector_type(16)));
typedef _Float16 half8  __attribute__((ext_vector_type(8)));
typedef float    f32x8  __attribute__((ext_vector_type(8)));
typedef float    f32x4  __attribute__((ext_vector_type(4)));

#define D_MODEL 1024
#define SEQ_L   2048
#define NHEAD   16
#define DHEAD   64
#define BATCH   4
#define M_TOT   (BATCH * SEQ_L)   // 8192
#define LSTR    40                // padded LDS row stride in halves (80B, 16B-aligned, conflict-free)

enum { OUT_F32 = 0, OUT_QK = 1, OUT_VT = 2 };

// Load a 16-half WMMA A/B fragment: lanes<16 take K{0..7,16..23}, lanes>=16 take K{8..15,24..31}
// (matches CDNA5 16-bit A/B VGPR layout). Works for global or LDS pointers.
__device__ __forceinline__ half16 load_frag16(const _Float16* rowbase, int klo) {
  half8 lo = *(const half8*)(rowbase + klo);
  half8 hi = *(const half8*)(rowbase + klo + 16);
  half16 r;
#pragma unroll
  for (int i = 0; i < 8; ++i) { r[i] = lo[i]; r[i + 8] = hi[i]; }
  return r;
}

__device__ __forceinline__ f32x8 wmma_f16(half16 a, half16 b, f32x8 c) {
  return __builtin_amdgcn_wmma_f32_16x16x32_f16(false, a, false, b, (short)0, c, false, false);
}

// ---------------- frequency-energy gate: gate[b,l] = (E/(maxE+1e-8) > 0.1) ? 1.0 : 0.1 ----
__global__ __launch_bounds__(256) void fsa_gate_kernel(const float* __restrict__ ec,
                                                       float* __restrict__ gate) {
  int b = blockIdx.x;
  __shared__ float ebuf[SEQ_L];
  __shared__ float red[256];
  float lmax = 0.f;
  for (int l = threadIdx.x; l < SEQ_L; l += 256) {
    const float* p = ec + ((size_t)b * SEQ_L + l) * 32;
    float s = 0.f;
#pragma unroll
    for (int f = 0; f < 32; ++f) s += fabsf(p[f]);
    float e = s * (1.0f / 32.0f);
    ebuf[l] = e;
    lmax = fmaxf(lmax, e);
  }
  red[threadIdx.x] = lmax;
  __syncthreads();
  for (int off = 128; off > 0; off >>= 1) {
    if (threadIdx.x < off) red[threadIdx.x] = fmaxf(red[threadIdx.x], red[threadIdx.x + off]);
    __syncthreads();
  }
  float mx = red[0] + 1e-8f;
  for (int l = threadIdx.x; l < SEQ_L; l += 256)
    gate[b * SEQ_L + l] = (ebuf[l] / mx > 0.1f) ? 1.0f : 0.1f;
}

// ---------------- WMMA GEMM: dst = A[8192,1024] x W[1024,1024] + bias ----------------------
// 256 threads = 8 waves; block tile 128(M) x 64(N); each wave owns a 32x32 sub-tile
// (4 WMMAs/k-step with A/B fragment reuse). K-step 32, f16 LDS staging, 32B global loads.
template <int A_HALF, int MODE>
__global__ __launch_bounds__(256) void fsa_gemm_wmma(const void* __restrict__ Asrc,
                                                     const float* __restrict__ Wsrc,
                                                     const float* __restrict__ bias,
                                                     void* __restrict__ dst) {
  __shared__ __align__(16) _Float16 Asm[128][LSTR];  // A rows     (M x K-chunk)
  __shared__ __align__(16) _Float16 Bsm[64][LSTR];   // W columns  (N x K-chunk) == B^T

  const int n0 = blockIdx.x * 64;
  const int m0 = blockIdx.y * 128;
  const int t    = threadIdx.x;
  const int wave = t >> 5;
  const int lane = t & 31;
  const int wm   = (wave >> 1) * 32;    // wave's M offset (0,32,64,96)
  const int wn   = (wave & 1) * 32;     // wave's N offset (0,32)
  const int lr   = lane & 15;
  const int klo  = (lane < 16) ? 0 : 8;

  f32x8 acc[2][2] = {};
  for (int kb = 0; kb < D_MODEL; kb += 32) {
    // --- stage A tile 128x32 as f16 (8 contiguous elements per thread, 2 row-groups) ---
    {
      const int r0 = t >> 2;            // 0..63
      const int c  = (t & 3) * 8;       // 0,8,16,24
#pragma unroll
      for (int i = 0; i < 2; ++i) {
        int r = r0 + i * 64;
        if (A_HALF) {
          *(half8*)&Asm[r][c] =
              *(const half8*)((const _Float16*)Asrc + (size_t)(m0 + r) * D_MODEL + kb + c);
        } else {
          const float* s = (const float*)Asrc + (size_t)(m0 + r) * D_MODEL + kb + c;
          f32x4 v0 = *(const f32x4*)s;
          f32x4 v1 = *(const f32x4*)(s + 4);
          half8 hh;
#pragma unroll
          for (int j = 0; j < 4; ++j) { hh[j] = (_Float16)v0[j]; hh[j + 4] = (_Float16)v1[j]; }
          *(half8*)&Asm[r][c] = hh;
        }
      }
    }
    // --- stage W tile 32(k)x64(n) transposed (coalesced global read, b16 LDS scatter) ---
    {
      const int k = t >> 3;             // 0..31
      const int c = (t & 7) * 8;        // 0..56
      const float* s = Wsrc + (size_t)(kb + k) * D_MODEL + n0 + c;
      f32x4 v0 = *(const f32x4*)s;
      f32x4 v1 = *(const f32x4*)(s + 4);
#pragma unroll
      for (int j = 0; j < 4; ++j) {
        Bsm[c + j][k]     = (_Float16)v0[j];
        Bsm[c + 4 + j][k] = (_Float16)v1[j];
      }
    }
    __syncthreads();
    half16 af0 = load_frag16(&Asm[wm + lr][0], klo);
    half16 af1 = load_frag16(&Asm[wm + 16 + lr][0], klo);
    half16 bf0 = load_frag16(&Bsm[wn + lr][0], klo);
    half16 bf1 = load_frag16(&Bsm[wn + 16 + lr][0], klo);
    acc[0][0] = wmma_f16(af0, bf0, acc[0][0]);
    acc[0][1] = wmma_f16(af0, bf1, acc[0][1]);
    acc[1][0] = wmma_f16(af1, bf0, acc[1][0]);
    acc[1][1] = wmma_f16(af1, bf1, acc[1][1]);
    __syncthreads();
  }

  const int rhalf = 8 * (lane >> 4);    // C layout: M = i + 8*(lane/16), N = lane%16
#pragma unroll
  for (int mt = 0; mt < 2; ++mt) {
#pragma unroll
    for (int nt = 0; nt < 2; ++nt) {
      int col = n0 + wn + nt * 16 + lr;
      float bv = bias[col];
#pragma unroll
      for (int i = 0; i < 8; ++i) {
        int row = m0 + wm + mt * 16 + i + rhalf;   // row in [0, 8192) == b*L + lseq
        float v = acc[mt][nt][i] + bv;
        if (MODE == OUT_F32) {
          ((float*)dst)[(size_t)row * D_MODEL + col] = v;
        } else {
          int b = row >> 11, lseq = row & (SEQ_L - 1);
          int h = col >> 6,  d    = col & (DHEAD - 1);
          if (MODE == OUT_QK)
            ((_Float16*)dst)[(((size_t)(b * NHEAD + h)) * SEQ_L + lseq) * DHEAD + d] = (_Float16)v;
          else  // OUT_VT : V stored transposed [B,H,Dh,L]
            ((_Float16*)dst)[(((size_t)(b * NHEAD + h)) * DHEAD + d) * SEQ_L + lseq] = (_Float16)v;
        }
      }
    }
  }
}

// ---------------- flash attention: 1 wave per (b,h, 32-query tile pair) ---------------------
// Two 16-query tiles per wave: every K fragment and every V fragment feeds two WMMAs.
__global__ __launch_bounds__(32) void fsa_attn_kernel(const _Float16* __restrict__ Q,
                                                      const _Float16* __restrict__ Km,
                                                      const _Float16* __restrict__ VT,
                                                      const float* __restrict__ gate,
                                                      _Float16* __restrict__ AO) {
  const int qt = blockIdx.x;            // 0..63 (32 queries each)
  const int bh = blockIdx.y;            // 0..63
  const int b = bh >> 4, h = bh & 15;
  const int lane = threadIdx.x;
  const int lr = lane & 15;
  const int klo = (lane < 16) ? 0 : 8;
  const int qbase = qt * 32;
  const int rbase = 8 * (lane >> 4);

  half16 qf[2][2];
#pragma unroll
  for (int ql = 0; ql < 2; ++ql) {
    const _Float16* qrow = Q + ((size_t)bh * SEQ_L + qbase + ql * 16 + lr) * DHEAD;
    qf[ql][0] = load_frag16(qrow, klo);        // d 0..31
    qf[ql][1] = load_frag16(qrow + 32, klo);   // d 32..63
  }

  f32x8 oacc[2][4] = {};
  float m_i[2][8], l_i[2][8];
#pragma unroll
  for (int ql = 0; ql < 2; ++ql)
#pragma unroll
    for (int i = 0; i < 8; ++i) { m_i[ql][i] = -1e30f; l_i[ql][i] = 0.f; }

  __shared__ __align__(16) _Float16 Pbuf[2][16 * LSTR];
  const float scale = 0.125f;   // 1/sqrt(64)

  for (int kb2 = 0; kb2 < SEQ_L / 32; ++kb2) {
    const int kbase = kb2 * 32;
    // K fragments straight from global (row-major [pos,64] == B^T layout for Q.K^T)
    const _Float16* krow0 = Km + ((size_t)bh * SEQ_L + kbase + lr) * DHEAD;
    const _Float16* krow1 = krow0 + 16 * DHEAD;
    half16 kf00 = load_frag16(krow0, klo);
    half16 kf01 = load_frag16(krow0 + 32, klo);
    half16 kf10 = load_frag16(krow1, klo);
    half16 kf11 = load_frag16(krow1 + 32, klo);

    const float g0 = gate[b * SEQ_L + kbase + lr];        // key col = lane%16 in C layout
    const float g1 = gate[b * SEQ_L + kbase + 16 + lr];

#pragma unroll
    for (int ql = 0; ql < 2; ++ql) {
      f32x8 s0 = {}, s1 = {};
      s0 = wmma_f16(qf[ql][0], kf00, s0);
      s0 = wmma_f16(qf[ql][1], kf01, s0);
      s1 = wmma_f16(qf[ql][0], kf10, s1);
      s1 = wmma_f16(qf[ql][1], kf11, s1);

#pragma unroll
      for (int i = 0; i < 8; ++i) {
        float v0 = s0[i] * scale * g0;
        float v1 = s1[i] * scale * g1;
        float mx = fmaxf(v0, v1);
        mx = fmaxf(mx, __shfl_xor(mx, 1));
        mx = fmaxf(mx, __shfl_xor(mx, 2));
        mx = fmaxf(mx, __shfl_xor(mx, 4));
        mx = fmaxf(mx, __shfl_xor(mx, 8));     // row max within 16-lane half
        float mnew = fmaxf(m_i[ql][i], mx);
        float a = __expf(m_i[ql][i] - mnew);
        float p0 = __expf(v0 - mnew);
        float p1 = __expf(v1 - mnew);
        float rs = p0 + p1;
        rs += __shfl_xor(rs, 1);
        rs += __shfl_xor(rs, 2);
        rs += __shfl_xor(rs, 4);
        rs += __shfl_xor(rs, 8);               // row sum
        l_i[ql][i] = l_i[ql][i] * a + rs;
        m_i[ql][i] = mnew;
#pragma unroll
        for (int dt = 0; dt < 4; ++dt) oacc[ql][dt][i] *= a;
        // restage P (C layout -> A layout) through LDS
        Pbuf[ql][(rbase + i) * LSTR + lr]      = (_Float16)p0;
        Pbuf[ql][(rbase + i) * LSTR + 16 + lr] = (_Float16)p1;
      }
    }
    __syncthreads();
    half16 pa0 = load_frag16(&Pbuf[0][lr * LSTR], klo);
    half16 pa1 = load_frag16(&Pbuf[1][lr * LSTR], klo);
    // V fragments from pre-transposed VT [B,H,Dh,L]; each feeds both query tiles
#pragma unroll
    for (int dt = 0; dt < 4; ++dt) {
      const _Float16* vrow = VT + ((size_t)bh * DHEAD + dt * 16 + lr) * SEQ_L + kbase;
      half16 vf = load_frag16(vrow, klo);
      oacc[0][dt] = wmma_f16(pa0, vf, oacc[0][dt]);
      oacc[1][dt] = wmma_f16(pa1, vf, oacc[1][dt]);
    }
    __syncthreads();
  }

#pragma unroll
  for (int ql = 0; ql < 2; ++ql) {
#pragma unroll
    for (int i = 0; i < 8; ++i) {
      float inv = (l_i[ql][i] > 0.f) ? 1.0f / l_i[ql][i] : 0.f;   // nan_to_num guard
#pragma unroll
      for (int dt = 0; dt < 4; ++dt) oacc[ql][dt][i] *= inv;
    }
#pragma unroll
    for (int dt = 0; dt < 4; ++dt) {
      int col = h * DHEAD + dt * 16 + lr;
#pragma unroll
      for (int i = 0; i < 8; ++i) {
        int row = qbase + ql * 16 + rbase + i;
        AO[((size_t)b * SEQ_L + row) * D_MODEL + col] = (_Float16)(oacc[ql][dt][i]);
      }
    }
  }
}

extern "C" void kernel_launch(void* const* d_in, const int* in_sizes, int n_in,
                              void* d_out, int out_size, void* d_ws, size_t ws_size,
                              hipStream_t stream) {
  const float* q_x = (const float*)d_in[0];
  const float* k_x = (const float*)d_in[1];
  const float* v_x = (const float*)d_in[2];
  const float* ec  = (const float*)d_in[3];
  const float* Wq  = (const float*)d_in[4];
  const float* bq  = (const float*)d_in[5];
  const float* Wk  = (const float*)d_in[6];
  const float* bk  = (const float*)d_in[7];
  const float* Wv  = (const float*)d_in[8];
  const float* bv  = (const float*)d_in[9];
  const float* Wo  = (const float*)d_in[10];
  const float* bo  = (const float*)d_in[11];

  char* ws = (char*)d_ws;
  const size_t HALF_BUF = (size_t)BATCH * NHEAD * SEQ_L * DHEAD * sizeof(_Float16); // 16 MB
  _Float16* Qb  = (_Float16*)(ws);
  _Float16* Kb  = (_Float16*)(ws + HALF_BUF);
  _Float16* VTb = (_Float16*)(ws + 2 * HALF_BUF);
  _Float16* AOb = (_Float16*)(ws + 3 * HALF_BUF);
  float*    gb  = (float*)(ws + 4 * HALF_BUF);

  fsa_gate_kernel<<<BATCH, 256, 0, stream>>>(ec, gb);

  dim3 ggrid(D_MODEL / 64, M_TOT / 128);   // (16, 64)
  fsa_gemm_wmma<0, OUT_QK><<<ggrid, 256, 0, stream>>>(q_x, Wq, bq, Qb);
  fsa_gemm_wmma<0, OUT_QK><<<ggrid, 256, 0, stream>>>(k_x, Wk, bk, Kb);
  fsa_gemm_wmma<0, OUT_VT><<<ggrid, 256, 0, stream>>>(v_x, Wv, bv, VTb);

  fsa_attn_kernel<<<dim3(SEQ_L / 32, BATCH * NHEAD), 32, 0, stream>>>(Qb, Kb, VTb, gb, AOb);

  fsa_gemm_wmma<1, OUT_F32><<<ggrid, 256, 0, stream>>>(AOb, Wo, bo, d_out);
}
